// KWS_LSTM_72035191488639
// MI455X (gfx1250) — matrix operand order repaired and checked by reading, not compiled, not run on hardware
//
#include <hip/hip_runtime.h>

typedef __attribute__((ext_vector_type(16))) _Float16 v16h;
typedef __attribute__((ext_vector_type(8)))  _Float16 v8h;
typedef __attribute__((ext_vector_type(2)))  _Float16 h2;
typedef __attribute__((ext_vector_type(8)))  float    v8f;

#define T_STEPS 256
#define BATCH   1024
#define IN_DIM  40
#define HID     256
#define GATES   (4*HID)
#define OUT_DIM 12
#define MTILE   32                 // batch rows per workgroup
#define NBLK    (BATCH/MTILE)      // 32 workgroups
#define HSTR    264                // 256 + 8 halfs pad -> conflict-free ds_load_b128
#define XSTR    72                 // 64 + 8 halfs pad
#define KIH     64                 // w_ih K padded 40 -> 64 in precomputed f16 buffer
#define NOISE_LEVEL 0.1f

// workspace layout (bytes): [0..7] wmax floats, [256..) qW_ih f16, then qW_hh f16
#define QIH_BYTE_OFF 256
#define QIH_BYTES    (GATES*KIH*2)          // 131072
#define QHH_BYTE_OFF (QIH_BYTE_OFF + QIH_BYTES)

// ---------------- helpers ----------------

// quant_clip forward value: round(clamp(x,0,1)*scale)/scale  (RNE == jnp.round)
__device__ __forceinline__ float qround(float x, float s, float inv) {
  return __builtin_rintf(__builtin_amdgcn_fmed3f(x, 0.f, 1.f) * s) * inv;  // v_med3+v_rndne
}
__device__ __forceinline__ float qcs(float x){ return qround(x,128.f,0.0078125f);  } // signed 8b
// for values already in [0,1] (sigmoid outputs): clamp is a no-op
__device__ __forceinline__ float qsig(float x){ return __builtin_rintf(x*256.f)*0.00390625f; }
__device__ __forceinline__ float sigmoidf(float x){ return 1.f/(1.f+__expf(-x)); }
__device__ __forceinline__ float ftanh(float x){ return 1.f - 2.f/(__expf(2.f*x)+1.f); }

__device__ __forceinline__ v8f wmma16(v16h a, v16h b, v8f c){
  // D = A(16x32 f16) * B(32x16 f16) + C(16x16 f32)
  return __builtin_amdgcn_wmma_f32_16x16x32_f16(false, a, false, b, (short)0, c, false, false);
}

// A fragment (16x32 f16) from row-major LDS tile.
// lanes 0-15: row M=lane, K 0..15; lanes 16-31: same rows, K 16..31
__device__ __forceinline__ v16h load_a(const _Float16* tile, int stride, int lane){
  int row = lane & 15;
  int kb  = (lane >> 4) * 16;
  const _Float16* p = tile + row*stride + kb;
  v16h a;
  #pragma unroll
  for (int j=0;j<8;++j) a[j]   = p[j];       // 16B contiguous -> ds_load_b128
  #pragma unroll
  for (int j=0;j<8;++j) a[8+j] = p[8+j];     // 16B contiguous -> ds_load_b128
  return a;
}

// B fragment (32x16 f16): pre-offset quantized f16 weights + packed-f16 hash noise.
// lane = column n (lanes 0-15 K=0..15, lanes 16-31 K=16..31), K contiguous per lane.
// Stored weights already include the noise affine offset c = -1.5*sqrt(12)*nscale,
// so each pair costs ONE v_pk_fma_f16: b = u*s + qw', u = mantissa-packed [1,2).
__device__ __forceinline__ v16h gen_b(const _Float16* __restrict__ qw, int ld,
                                      int ncol, int kbase, int lane,
                                      h2 s, unsigned base0){
  const int n  = ncol + (lane & 15);
  const int k0 = kbase + (lane >> 4) * 16;
  const v8h* p = (const v8h*)(qw + (size_t)n * ld + k0);   // 16B aligned
  v8h lo = p[0], hi = p[1];                                // 2x global_load_b128
  unsigned base = base0 + (unsigned)(n * 256 + (k0 >> 2));
  v16h b;
  #pragma unroll
  for (int q=0; q<4; ++q){
    unsigned v = (base + (unsigned)q) * 0x9E3779B1u;
    v ^= v >> 16;
    unsigned v2 = v * 0x85EBCA6Bu;
    unsigned ub0 = (v  & 0x03FF03FFu) | 0x3C003C00u;       // two f16 in [1,2)
    unsigned ub1 = (v2 & 0x03FF03FFu) | 0x3C003C00u;       // two more
    h2 w0, w1;
    if (q < 2){
      w0[0]=lo[4*q];   w0[1]=lo[4*q+1];
      w1[0]=lo[4*q+2]; w1[1]=lo[4*q+3];
    } else {
      int qq = q-2;
      w0[0]=hi[4*qq];   w0[1]=hi[4*qq+1];
      w1[0]=hi[4*qq+2]; w1[1]=hi[4*qq+3];
    }
    h2 r0 = __builtin_bit_cast(h2, ub0) * s + w0;          // single pk_fma per pair
    h2 r1 = __builtin_bit_cast(h2, ub1) * s + w1;
    b[4*q]   = r0[0]; b[4*q+1] = r0[1];
    b[4*q+2] = r1[0]; b[4*q+3] = r1[1];
  }
  return b;
}

// ---------------- pre-pass kernels ----------------

__global__ void ws_init(float* ws){ if (threadIdx.x < 2) ws[threadIdx.x] = 0.f; }

__global__ void max_reduce(const float* __restrict__ x, int n, float* slot){
  __shared__ float sm[256];
  float m = 0.f;
  for (int i = blockIdx.x*256 + threadIdx.x; i < n; i += gridDim.x*256)
    m = __builtin_fmaxf(m, x[i]);
  sm[threadIdx.x] = m;
  __syncthreads();
  for (int s=128; s>0; s>>=1){
    if (threadIdx.x < s) sm[threadIdx.x] = __builtin_fmaxf(sm[threadIdx.x], sm[threadIdx.x+s]);
    __syncthreads();
  }
  if (threadIdx.x==0) atomicMax((int*)slot, __float_as_int(sm[0]));  // max > 0 here
}

// quantize weights once (time-invariant) AND fold in the noise affine offset:
//   q[n][k] = quant_clip(w[n][k]) - 1.5*sqrt(12)*NOISE_LEVEL*wmax   (pad: offset only)
__global__ void quant_w(const float* __restrict__ w, _Float16* __restrict__ q,
                        int rows, int kin, int kout,
                        const float* __restrict__ wmax, int slot){
  const float cadd = -5.1961524f * NOISE_LEVEL * wmax[slot];
  int ntot = rows * kout;
  for (int i = blockIdx.x*256 + threadIdx.x; i < ntot; i += gridDim.x*256){
    int n = i / kout, k = i - n*kout;
    float v = (k < kin) ? qcs(w[(size_t)n*kin + k]) : 0.f;
    q[i] = (_Float16)(v + cadd);
  }
}

// ---------------- fused persistent LSTM ----------------

__global__ __launch_bounds__(256)
void lstm_fused(const float* __restrict__ inputs,
                const float* __restrict__ b_ih,
                const float* __restrict__ b_hh,
                const float* __restrict__ out_w,
                const float* __restrict__ out_b,
                const float* __restrict__ wmax,
                const _Float16* __restrict__ qih,
                const _Float16* __restrict__ qhh,
                float* __restrict__ out)
{
  __shared__ _Float16 h_lds[2][MTILE][HSTR];   // double-buffered recurrent state (f16)
  __shared__ _Float16 x_lds[MTILE][XSTR];      // quantized x_t, K padded 40->64

  const int tid  = threadIdx.x;
  const int lane = tid & 31;
  const int wv   = tid >> 5;                   // 8 waves; wave owns unit tiles {2w, 2w+1}
  const int b0   = blockIdx.x * MTILE;
  const v8f vzero = {};

  // packed-f16 noise slope: n = u * (sqrt(12)*nscale) + (offset prefolded into qw)
  const float ns_ih = wmax[0] * NOISE_LEVEL;
  const float ns_hh = wmax[1] * NOISE_LEVEL;
  h2 s_ih, s_hh;
  s_ih[0] = s_ih[1] = (_Float16)(3.4641016f * ns_ih);
  s_hh[0] = s_hh[1] = (_Float16)(3.4641016f * ns_hh);

  for (int i = tid; i < MTILE*HSTR; i += 256)
    (&h_lds[0][0][0])[i] = (_Float16)0.f;

  float cst[2][2][8];                          // cell state lives in VGPRs forever
  #pragma unroll
  for (int a=0;a<2;++a)
    #pragma unroll
    for (int m=0;m<2;++m)
      #pragma unroll
      for (int r=0;r<8;++r) cst[a][m][r]=0.f;

  float bias[2][4];                            // qb_ih[n]+qb_hh[n], per (u-tile, gate)
  #pragma unroll
  for (int ui=0; ui<2; ++ui){
    int u = wv*2 + ui;
    #pragma unroll
    for (int g=0; g<4; ++g){
      int n = g*HID + u*16 + (lane & 15);
      bias[ui][g] = qcs(b_ih[n]) + qcs(b_hh[n]);
    }
  }

  int cur = 0;
  for (int t = 0; t < T_STEPS; ++t){
    // stage quantized x_t for this batch tile
    for (int i = tid; i < MTILE*64; i += 256){
      int r = i >> 6, c = i & 63;
      float v = 0.f;
      if (c < IN_DIM)
        v = qcs(inputs[((size_t)t*BATCH + b0 + r)*IN_DIM + c]);
      x_lds[r][c] = (_Float16)v;
    }
    // prefetch next timestep's x slice (gfx1250 global_prefetch_b8)
    if (t + 1 < T_STEPS)
      __builtin_prefetch(inputs + ((size_t)(t+1)*BATCH + b0)*IN_DIM + tid*5, 0, 1);
    __syncthreads();

    const int nxt = cur ^ 1;
    const unsigned base_ih = (unsigned)(t*2+1) * 0x27D4EB2Fu;
    const unsigned base_hh = (unsigned)(t*2+2) * 0x27D4EB2Fu;

    #pragma unroll
    for (int ui=0; ui<2; ++ui){
      int u = wv*2 + ui;
      v8f acc[4][2];
      #pragma unroll
      for (int g=0; g<4; ++g){ acc[g][0] = vzero; acc[g][1] = vzero; }

      // x_t @ (qW_ih + noise): K padded to 64 (pad rows of B hit zero A cols -> no mask)
      #pragma unroll
      for (int ks=0; ks<2; ++ks){
        v16h a0 = load_a(&x_lds[0][ks*32],  XSTR, lane);
        v16h a1 = load_a(&x_lds[16][ks*32], XSTR, lane);
        #pragma unroll
        for (int g=0; g<4; ++g){
          v16h b = gen_b(qih, KIH, g*HID + u*16, ks*32, lane, s_ih, base_ih);
          acc[g][0] = wmma16(a0, b, acc[g][0]);
          acc[g][1] = wmma16(a1, b, acc[g][1]);
        }
      }
      // h @ (qW_hh + noise): K = 256
      for (int ks=0; ks<8; ++ks){
        v16h a0 = load_a(&h_lds[cur][0][ks*32],  HSTR, lane);
        v16h a1 = load_a(&h_lds[cur][16][ks*32], HSTR, lane);
        #pragma unroll
        for (int g=0; g<4; ++g){
          v16h b = gen_b(qhh, HID, g*HID + u*16, ks*32, lane, s_hh, base_hh);
          acc[g][0] = wmma16(a0, b, acc[g][0]);
          acc[g][1] = wmma16(a1, b, acc[g][1]);
        }
      }

      // element-wise LSTM cell — C/D layout keeps i/f/g/o of one (row,unit) in one lane
      const int colw  = u*16 + (lane & 15);
      const int rbase = (lane >> 4) * 8;
      #pragma unroll
      for (int m=0;m<2;++m){
        #pragma unroll
        for (int r=0;r<8;++r){
          float iv = qsig(sigmoidf(acc[0][m][r] + bias[ui][0]));
          float fv = qsig(sigmoidf(acc[1][m][r] + bias[ui][1]));
          float gv = qcs (ftanh   (acc[2][m][r] + bias[ui][2]));
          float ov = qsig(sigmoidf(acc[3][m][r] + bias[ui][3]));
          float cy = qcs(fv*cst[ui][m][r] + iv*gv);
          float hy = qcs(ov*ftanh(cy));
          cst[ui][m][r] = cy;
          h_lds[nxt][m*16 + rbase + r][colw] = (_Float16)hy;
        }
      }
    }
    __syncthreads();
    cur ^= 1;
  }

  // tiny output projection: h_T[32,256] @ out_w.T[256,12]
  for (int i = tid; i < MTILE*OUT_DIM; i += 256){
    int r = i / OUT_DIM, o = i % OUT_DIM;
    float s = out_b[o];
    const float* wr = out_w + (size_t)o*HID;
    #pragma unroll 4
    for (int j=0;j<HID;++j) s += (float)h_lds[cur][r][j] * wr[j];
    out[(size_t)(b0+r)*OUT_DIM + o] = qsig(sigmoidf(s));
  }
}

// ---------------- launch ----------------

extern "C" void kernel_launch(void* const* d_in, const int* in_sizes, int n_in,
                              void* d_out, int out_size, void* d_ws, size_t ws_size,
                              hipStream_t stream){
  const float* inputs = (const float*)d_in[0];
  const float* w_ih   = (const float*)d_in[1];
  const float* w_hh   = (const float*)d_in[2];
  const float* b_ih   = (const float*)d_in[3];
  const float* b_hh   = (const float*)d_in[4];
  const float* out_w  = (const float*)d_in[5];
  const float* out_b  = (const float*)d_in[6];
  float*     ws  = (float*)d_ws;
  _Float16*  qih = (_Float16*)((char*)d_ws + QIH_BYTE_OFF);
  _Float16*  qhh = (_Float16*)((char*)d_ws + QHH_BYTE_OFF);
  float*     out = (float*)d_out;

  ws_init<<<1, 32, 0, stream>>>(ws);
  max_reduce<<<40, 256, 0, stream>>>(w_ih, GATES*IN_DIM, ws + 0);
  max_reduce<<<64, 256, 0, stream>>>(w_hh, GATES*HID,    ws + 1);
  quant_w<<<64, 256, 0, stream>>>(w_ih, qih, GATES, IN_DIM, KIH, ws, 0);
  quant_w<<<128, 256, 0, stream>>>(w_hh, qhh, GATES, HID, HID, ws, 1);
  lstm_fused<<<NBLK, 256, 0, stream>>>(inputs, b_ih, b_hh, out_w, out_b,
                                       ws, qih, qhh, out);
}
